// DGN_43516608643755
// MI455X (gfx1250) — matrix operand order repaired and checked by reading, not compiled, not run on hardware
//
#include <hip/hip_runtime.h>

// ---------------------------------------------------------------------------
// DGN forward for MI455X (gfx1250): all GEMMs via v_wmma_f32_16x16x32_bf16.
// ---------------------------------------------------------------------------

typedef __bf16 bf16_t;
typedef __attribute__((ext_vector_type(16))) __bf16 v16bf;
typedef __attribute__((ext_vector_type(8)))  __bf16 v8bf;
typedef __attribute__((ext_vector_type(8)))  float  v8f;

#define WMMA_BF16(a, b, c)                                                   \
  __builtin_amdgcn_wmma_f32_16x16x32_bf16(false, (a), false, (b), (short)0,  \
                                          (c), false, false)

static constexpr int BATCH = 16;
static constexpr int NODES = 1024;
static constexpr int OBS   = 64;
static constexpr int HID   = 128;
static constexpr int ACT   = 32;
static constexpr int MROWS = BATCH * NODES;   // 16384

// ---------------------------------------------------------------------------
// WMMA fragment helpers (wave32). Layouts used here (contiguous K, same in A and B):
//  A (16x32 bf16): lane m=lane&15, K = (lane<16?0:16)+0..15,
//                  -> two contiguous 16B loads.
//  B (32x16 bf16): lane n=lane&15, K = (lane<16?0:16)+0..15 of column n;
//                  with Bt stored row-major [N][K] this is one 32B load.
//  C/D (16x16 f32): lane n=lane&15, rows m0..m0+7 with m0=(lane<16?0:8).
// ---------------------------------------------------------------------------
__device__ __forceinline__ int lane_id() { return (int)(threadIdx.x & 31u); }

__device__ __forceinline__ v16bf load_a_frag(const bf16_t* A, int lda) {
  int lane = lane_id();
  int m    = lane & 15;
  int k0   = (lane >> 4) << 4;                 // 0 or 16
  const bf16_t* p = A + (size_t)m * lda + k0;
  union { v16bf v; v8bf h[2]; } u;
  u.h[0] = *(const v8bf*)(p);                  // K = k0 .. k0+7
  u.h[1] = *(const v8bf*)(p + 8);              // K = k0+8 .. k0+15
  return u.v;
}

__device__ __forceinline__ v16bf load_b_frag(const bf16_t* Bt, int ldb) {
  int lane = lane_id();
  int n    = lane & 15;
  int k0   = (lane >> 4) << 4;                 // 0 or 16
  return *(const v16bf*)(Bt + (size_t)n * ldb + k0);
}

// ---------------------------------------------------------------------------
// Generic GEMM: C[M,N] = act(A[M,K] @ Bt[N,K]^T + bias)
//   one wave per 16x16 tile, K-loop in steps of 32.
//   OUT_F32:   write f32 row-major (final Q-net)
//   OUT_TRANS: write bf16 transposed per-batch ([b][n][node]) for v^T
// ---------------------------------------------------------------------------
template <bool RELU, bool OUT_F32, bool OUT_TRANS>
__global__ __launch_bounds__(256) void gemm_bias_kernel(
    const bf16_t* __restrict__ A, int lda,
    const bf16_t* __restrict__ Bt, int ldb,
    const float* __restrict__ bias,
    void* __restrict__ Cout, int ldc, int M, int N, int K) {
  int wave    = (int)((blockIdx.x * blockDim.x + threadIdx.x) >> 5);
  int tiles_n = N >> 4;
  int tiles   = (M >> 4) * tiles_n;
  if (wave >= tiles) return;                    // wave-uniform: EXEC stays full
  int tm = wave / tiles_n;
  int tn = wave - tm * tiles_n;

  const bf16_t* Ap = A + (size_t)tm * 16 * lda;
  const bf16_t* Bp = Bt + (size_t)tn * 16 * ldb;

  v8f acc = {};
  for (int k = 0; k < K; k += 32) {
    v16bf af = load_a_frag(Ap + k, lda);
    v16bf bf = load_b_frag(Bp + k, ldb);
    acc = WMMA_BF16(af, bf, acc);
  }

  int lane = lane_id();
  int n    = tn * 16 + (lane & 15);
  int m0   = tm * 16 + ((lane >> 4) << 3);
  float bv = bias[n];
#pragma unroll
  for (int r = 0; r < 8; ++r) {
    float v = acc[r] + bv;
    if (RELU) v = fmaxf(v, 0.f);
    int m = m0 + r;
    if (OUT_F32) {
      ((float*)Cout)[(size_t)m * ldc + n] = v;
    } else if (OUT_TRANS) {
      int bb = m >> 10, node = m & 1023;        // per-batch [HID][NODES]
      ((bf16_t*)Cout)[((size_t)bb * HID + n) * NODES + node] = (bf16_t)v;
    } else {
      ((bf16_t*)Cout)[(size_t)m * ldc + n] = (bf16_t)v;
    }
  }
}

// ---------------------------------------------------------------------------
// Fused attention: per workgroup = (batch b, 16-row q tile).
//  P1: s1 = q @ k^T            -> LDS bf16 [16][1024]
//  P2: s2 = s1 @ mask          -> LDS f32  [16][1024]   (Bt = maskT)
//  P3: masked softmax          -> LDS bf16 [16][1024]   (reuses s1 buffer)
//  P4: out = att @ v * 1/sqrt(H) -> LDS bf16 [16][128]  (Bt = vT)
//  P5: h' = relu(out @ Wo + bo)  -> global bf16
// ---------------------------------------------------------------------------
__global__ __launch_bounds__(128) void att_fused_kernel(
    const bf16_t* __restrict__ qb,     // [B*N, HID] bf16
    const bf16_t* __restrict__ kb,     // [B*N, HID] bf16
    const bf16_t* __restrict__ vT,     // [B][HID][N] bf16
    const bf16_t* __restrict__ maskT,  // [B][n][k] bf16 (transposed mask)
    const float*  __restrict__ maskf,  // [B][q][n] f32 (original mask)
    const bf16_t* __restrict__ WoT,    // [HID][HID] bf16
    const float*  __restrict__ bo,     // [HID]
    bf16_t* __restrict__ hout)         // [B*N, HID] bf16
{
  extern __shared__ char smem[];
  bf16_t* sA = (bf16_t*)smem;                              // 32 KB
  float*  sS = (float*)(smem + 16 * NODES * 2);            // 64 KB
  bf16_t* sO = (bf16_t*)(smem + 16 * NODES * 2 + 16 * NODES * 4);  // 4 KB

  int wg   = (int)blockIdx.x;
  int b    = wg >> 6;                 // / (NODES/16)
  int qt   = wg & 63;
  int tid  = (int)threadIdx.x;
  int wid  = tid >> 5;                // 0..3
  int lane = tid & 31;

  // Preload q A-fragments (K = 128 -> 4 frags), reused for all 64 n-tiles.
  const bf16_t* qp = qb + (size_t)(b * NODES + qt * 16) * HID;
  v16bf qa[4];
#pragma unroll
  for (int kk = 0; kk < 4; ++kk) qa[kk] = load_a_frag(qp + kk * 32, HID);

  // ---- Phase 1: s1 = q @ k^T ----
  const bf16_t* kbase = kb + (size_t)b * NODES * HID;
  for (int nt = wid * 16; nt < wid * 16 + 16; ++nt) {
    v8f acc = {};
    const bf16_t* Bp = kbase + (size_t)nt * 16 * HID;
#pragma unroll
    for (int kk = 0; kk < 4; ++kk)
      acc = WMMA_BF16(qa[kk], load_b_frag(Bp + kk * 32, HID), acc);
    int n = nt * 16 + (lane & 15), m0 = (lane >> 4) << 3;
#pragma unroll
    for (int r = 0; r < 8; ++r) sA[(size_t)(m0 + r) * NODES + n] = (bf16_t)acc[r];
  }
  __syncthreads();

  // ---- Phase 2: s2 = s1 @ mask (B-fragments from transposed mask) ----
  const bf16_t* mTb = maskT + (size_t)b * NODES * NODES;
  for (int nt = wid * 16; nt < wid * 16 + 16; ++nt) {
    v8f acc = {};
    const bf16_t* Bp = mTb + (size_t)nt * 16 * NODES;
    for (int k = 0; k < NODES; k += 32) {
      __builtin_prefetch(Bp + k + 256, 0, 1);   // global_prefetch_b8
      v16bf af = load_a_frag(sA + k, NODES);
      v16bf bf = load_b_frag(Bp + k, NODES);
      acc = WMMA_BF16(af, bf, acc);
    }
    int n = nt * 16 + (lane & 15), m0 = (lane >> 4) << 3;
#pragma unroll
    for (int r = 0; r < 8; ++r) sS[(size_t)(m0 + r) * NODES + n] = acc[r];
  }
  __syncthreads();

  // ---- Phase 3: masked softmax over n, write att (bf16) into sA ----
  {
    int row = tid >> 3;                 // 16 rows x 8 threads
    int sub = tid & 7;
    const float* mrow = maskf + ((size_t)(b * NODES + qt * 16 + row)) * NODES;
    float mx = -3.0e38f;
    for (int c = sub; c < NODES; c += 8)
      if (mrow[c] > 0.f) mx = fmaxf(mx, sS[(size_t)row * NODES + c]);
#pragma unroll
    for (int o = 4; o; o >>= 1) mx = fmaxf(mx, __shfl_xor(mx, o, 8));
    float sum = 0.f;
    for (int c = sub; c < NODES; c += 8) {
      float e = 0.f;
      if (mrow[c] > 0.f) e = __expf(sS[(size_t)row * NODES + c] - mx);
      sS[(size_t)row * NODES + c] = e;
      sum += e;
    }
#pragma unroll
    for (int o = 4; o; o >>= 1) sum += __shfl_xor(sum, o, 8);
    float inv = 1.f / sum;              // self-loop guarantees sum > 0
    for (int c = sub; c < NODES; c += 8)
      sA[(size_t)row * NODES + c] = (bf16_t)(sS[(size_t)row * NODES + c] * inv);
  }
  __syncthreads();

  // ---- Phase 4: out = (att @ v) / sqrt(HID) -> sO bf16 [16][128] ----
  const bf16_t* vTb = vT + (size_t)b * HID * NODES;
  const float scale = 0.08838834764831845f;     // 1/sqrt(128)
  for (int nt = wid * 2; nt < wid * 2 + 2; ++nt) {
    v8f acc = {};
    const bf16_t* Bp = vTb + (size_t)nt * 16 * NODES;
    for (int k = 0; k < NODES; k += 32)
      acc = WMMA_BF16(load_a_frag(sA + k, NODES), load_b_frag(Bp + k, NODES), acc);
    int n = nt * 16 + (lane & 15), m0 = (lane >> 4) << 3;
#pragma unroll
    for (int r = 0; r < 8; ++r) sO[(m0 + r) * HID + n] = (bf16_t)(acc[r] * scale);
  }
  __syncthreads();

  // ---- Phase 5: h' = relu(out @ Wo + bo) ----
  bf16_t* outp = hout + (size_t)(b * NODES + qt * 16) * HID;
  for (int nt = wid * 2; nt < wid * 2 + 2; ++nt) {
    v8f acc = {};
    const bf16_t* Bp = WoT + (size_t)nt * 16 * HID;
#pragma unroll
    for (int kk = 0; kk < 4; ++kk)
      acc = WMMA_BF16(load_a_frag(sO + kk * 32, HID), load_b_frag(Bp + kk * 32, HID), acc);
    int n = nt * 16 + (lane & 15), m0 = (lane >> 4) << 3;
    float bv = bo[n];
#pragma unroll
    for (int r = 0; r < 8; ++r)
      outp[(size_t)(m0 + r) * HID + n] = (bf16_t)fmaxf(acc[r] + bv, 0.f);
  }
}

// ---------------------------------------------------------------------------
// Conversion / layout kernels
// ---------------------------------------------------------------------------
__global__ void f32_to_bf16_kernel(const float* __restrict__ src,
                                   bf16_t* __restrict__ dst, long n) {
  long i = (long)blockIdx.x * blockDim.x + threadIdx.x;
  if (i < n) dst[i] = (bf16_t)src[i];
}

// dst[b][c][r] = src[b][r][c]  (batched transpose + convert to bf16)
__global__ void transpose_f32_to_bf16_kernel(const float* __restrict__ src,
                                             bf16_t* __restrict__ dst,
                                             int R, int C, long total) {
  long i = (long)blockIdx.x * blockDim.x + threadIdx.x;
  if (i >= total) return;
  long rc  = (long)R * C;
  long b   = i / rc;
  long rem = i - b * rc;
  int  c   = (int)(rem / R);
  int  r   = (int)(rem - (long)c * R);
  dst[i] = (bf16_t)src[b * rc + (long)r * C + c];
}

// ---------------------------------------------------------------------------
// Host launcher
// ---------------------------------------------------------------------------
extern "C" void kernel_launch(void* const* d_in, const int* in_sizes, int n_in,
                              void* d_out, int out_size, void* d_ws, size_t ws_size,
                              hipStream_t stream) {
  (void)in_sizes; (void)n_in; (void)out_size; (void)ws_size;

  const float* x    = (const float*)d_in[0];   // [16,1024,64]
  const float* mask = (const float*)d_in[1];   // [16,1024,1024]
  const float* We   = (const float*)d_in[2];   // [64,128]
  const float* be   = (const float*)d_in[3];
  // per layer dict order: Wv, bv, Wk, bk, Wq, bq, Wo, bo
  const float *aW[2][4], *ab[2][4];            // [layer][v,k,q,o]
  for (int l = 0; l < 2; ++l) {
    int base = 4 + l * 8;
    aW[l][0] = (const float*)d_in[base + 0]; ab[l][0] = (const float*)d_in[base + 1];
    aW[l][1] = (const float*)d_in[base + 2]; ab[l][1] = (const float*)d_in[base + 3];
    aW[l][2] = (const float*)d_in[base + 4]; ab[l][2] = (const float*)d_in[base + 5];
    aW[l][3] = (const float*)d_in[base + 6]; ab[l][3] = (const float*)d_in[base + 7];
  }
  const float* Wqn = (const float*)d_in[20];   // [128,32]
  const float* bqn = (const float*)d_in[21];
  float* out = (float*)d_out;                  // [16,1024,32]

  // ---- workspace carving (256B aligned) ----
  char* w = (char*)d_ws;
  auto carve = [&](size_t bytes) -> char* {
    char* p = w; w += (bytes + 255) & ~(size_t)255; return p;
  };
  bf16_t* xb    = (bf16_t*)carve((size_t)MROWS * OBS * 2);
  bf16_t* maskT = (bf16_t*)carve((size_t)BATCH * NODES * NODES * 2);
  bf16_t* WeT   = (bf16_t*)carve((size_t)OBS * HID * 2);
  bf16_t* WT[2][4];
  for (int l = 0; l < 2; ++l)
    for (int j = 0; j < 4; ++j) WT[l][j] = (bf16_t*)carve((size_t)HID * HID * 2);
  bf16_t* WqnT = (bf16_t*)carve((size_t)HID * ACT * 2);
  bf16_t* h1   = (bf16_t*)carve((size_t)MROWS * HID * 2);
  bf16_t* h2   = (bf16_t*)carve((size_t)MROWS * HID * 2);
  bf16_t* h3   = (bf16_t*)carve((size_t)MROWS * HID * 2);
  bf16_t* qbuf = (bf16_t*)carve((size_t)MROWS * HID * 2);
  bf16_t* kbuf = (bf16_t*)carve((size_t)MROWS * HID * 2);
  bf16_t* vT   = (bf16_t*)carve((size_t)MROWS * HID * 2);

  // ---- precision / layout prep ----
  { long n = (long)MROWS * OBS;
    f32_to_bf16_kernel<<<(n + 255) / 256, 256, 0, stream>>>(x, xb, n); }
  { long t = (long)OBS * HID;
    transpose_f32_to_bf16_kernel<<<(t + 255) / 256, 256, 0, stream>>>(We, WeT, OBS, HID, t); }
  for (int l = 0; l < 2; ++l)
    for (int j = 0; j < 4; ++j) {
      long t = (long)HID * HID;
      transpose_f32_to_bf16_kernel<<<(t + 255) / 256, 256, 0, stream>>>(aW[l][j], WT[l][j], HID, HID, t);
    }
  { long t = (long)HID * ACT;
    transpose_f32_to_bf16_kernel<<<(t + 255) / 256, 256, 0, stream>>>(Wqn, WqnT, HID, ACT, t); }
  { long t = (long)BATCH * NODES * NODES;
    transpose_f32_to_bf16_kernel<<<(t + 255) / 256, 256, 0, stream>>>(mask, maskT, NODES, NODES, t); }

  auto blocks_for = [](int M, int N) { return (((M >> 4) * (N >> 4)) + 7) / 8; };

  // ---- encoder: h1 = relu(x @ We + be) ----
  gemm_bias_kernel<true, false, false><<<blocks_for(MROWS, HID), 256, 0, stream>>>(
      xb, OBS, WeT, OBS, be, (void*)h1, HID, MROWS, HID, OBS);

  // ---- attention layers ----
  const size_t SMEM = (size_t)16 * NODES * 2 + (size_t)16 * NODES * 4 + (size_t)16 * HID * 2;
  bf16_t* hin = h1;
  bf16_t* houts[2] = {h2, h3};
  for (int l = 0; l < 2; ++l) {
    gemm_bias_kernel<true, false, false><<<blocks_for(MROWS, HID), 256, 0, stream>>>(
        hin, HID, WT[l][2], HID, ab[l][2], (void*)qbuf, HID, MROWS, HID, HID);  // q
    gemm_bias_kernel<true, false, false><<<blocks_for(MROWS, HID), 256, 0, stream>>>(
        hin, HID, WT[l][1], HID, ab[l][1], (void*)kbuf, HID, MROWS, HID, HID);  // k
    gemm_bias_kernel<true, false, true><<<blocks_for(MROWS, HID), 256, 0, stream>>>(
        hin, HID, WT[l][0], HID, ab[l][0], (void*)vT, NODES, MROWS, HID, HID);  // v^T
    att_fused_kernel<<<BATCH * (NODES / 16), 128, SMEM, stream>>>(
        qbuf, kbuf, vT, maskT, mask, WT[l][3], ab[l][3], houts[l]);
    hin = houts[l];
  }

  // ---- Q-net: out = h3 @ Wqn + bqn (f32, no activation) ----
  gemm_bias_kernel<false, true, false><<<blocks_for(MROWS, ACT), 256, 0, stream>>>(
      h3, HID, WqnT, HID, bqn, (void*)out, ACT, MROWS, ACT, HID);
}
